// CommBlock_49108656062925
// MI455X (gfx1250) — matrix operand order repaired and checked by reading, not compiled, not run on hardware
//
#include <hip/hip_runtime.h>
#include <hip/hip_bf16.h>

#define NNODES 40000
#define NEDGES 640000
#define FDIM   128
#define NHEADS 8
#define DHEAD  16
#define NEG_SLOPE 0.2f
#define LN_EPS 1e-5f

typedef __attribute__((ext_vector_type(16))) _Float16 v16h;
typedef __attribute__((ext_vector_type(8)))  _Float16 v8h;
typedef __attribute__((ext_vector_type(8)))  float    v8f;

// ---------------- small utility kernels ----------------

__global__ void k_to_f16(const float* __restrict__ in, _Float16* __restrict__ out, int n) {
    int i = blockIdx.x * 256 + threadIdx.x;
    if (i < n) out[i] = (_Float16)in[i];
}

// W is [K=128, N=128] row-major (x @ W). Produce wT[n][k] in f16 so WMMA B loads are contiguous.
__global__ void k_transpose_w(const float* __restrict__ w, _Float16* __restrict__ wT) {
    int n = blockIdx.x, k = threadIdx.x;
    wT[n * FDIM + k] = (_Float16)w[k * FDIM + n];
}

__global__ void k_fill(float* __restrict__ p, float v, int n) {
    int i = blockIdx.x * 256 + threadIdx.x;
    if (i < n) p[i] = v;
}

// ---------------- fused dual GEMM: xl = x@Wl, xr = x@Wr ----------------
// grid.x = NNODES/16 (M tiles), block = 256 (8 waves); wave w handles N-tile w (16 cols).
// A (16x32 f16): lane-half h holds K = 16h..16h+15 contiguous (matches B).
// B (32x16 f16): lane = column, half h holds K = 16h..16h+15 contiguous in wT[n][k].
__global__ __launch_bounds__(256) void k_gemm_xlxr(
    const _Float16* __restrict__ xh,   // [N,128] f16 row-major
    const _Float16* __restrict__ wlT,  // [128(out),128(in)] f16
    const _Float16* __restrict__ wrT,
    float* __restrict__ xl, float* __restrict__ xr)
{
    const int wave   = threadIdx.x >> 5;
    const int lane   = threadIdx.x & 31;
    const int half   = lane >> 4;
    const int lane16 = lane & 15;
    const int m0 = blockIdx.x * 16;
    const int n0 = wave * 16;

    const _Float16* arow  = xh  + (size_t)(m0 + lane16) * FDIM;
    const _Float16* blrow = wlT + (size_t)(n0 + lane16) * FDIM;
    const _Float16* brrow = wrT + (size_t)(n0 + lane16) * FDIM;

    v8f cl = {}; v8f cr = {};
#pragma unroll
    for (int k0 = 0; k0 < FDIM; k0 += 32) {
        union { v16h v; v8h h2[2]; } a;
        a.h2[0] = *(const v8h*)(arow + k0 + 16 * half);
        a.h2[1] = *(const v8h*)(arow + k0 + 16 * half + 8);
        v16h bl = *(const v16h*)(blrow + k0 + 16 * half);
        v16h br = *(const v16h*)(brrow + k0 + 16 * half);
        cl = __builtin_amdgcn_wmma_f32_16x16x32_f16(false, a.v, false, bl, (short)0, cl, false, false);
        cr = __builtin_amdgcn_wmma_f32_16x16x32_f16(false, a.v, false, br, (short)0, cr, false, false);
    }

    const int col   = n0 + lane16;
    const int rbase = m0 + 8 * half;
#pragma unroll
    for (int r = 0; r < 8; ++r) {
        xl[(size_t)(rbase + r) * FDIM + col] = cl[r];
        xr[(size_t)(rbase + r) * FDIM + col] = cr[r];
    }
}

// ---------------- edge kernels ----------------

__device__ __forceinline__ void atomicMaxF(float* addr, float v) {
    // monotonic int mapping: works mixed-sign with init = -inf
    if (v >= 0.0f) atomicMax((int*)addr, __float_as_int(v));
    else           atomicMin((unsigned int*)addr, __float_as_uint(v));
}

// e[eid,h] = dot(leakyrelu(xl[src]+xr[dst]), att[h]); atomic segment-max into emax[dst,h]
__global__ void k_edge_score(const float* __restrict__ xl, const float* __restrict__ xr,
                             const float* __restrict__ att,
                             const int* __restrict__ src, const int* __restrict__ dst,
                             float* __restrict__ e, float* __restrict__ emax)
{
    int idx = blockIdx.x * 256 + threadIdx.x;       // over E*H
    if (idx >= NEDGES * NHEADS) return;
    int eid = idx >> 3, hh = idx & 7;
    int s = src[eid], d = dst[eid];
    const float* pl = xl + (size_t)s * FDIM + hh * DHEAD;
    const float* pr = xr + (size_t)d * FDIM + hh * DHEAD;
    const float* pa = att + hh * DHEAD;
    float acc = 0.f;
#pragma unroll
    for (int i = 0; i < DHEAD; ++i) {
        float m = pl[i] + pr[i];
        m = (m > 0.f) ? m : NEG_SLOPE * m;
        acc += m * pa[i];
    }
    e[idx] = acc;
    atomicMaxF(emax + (size_t)d * NHEADS + hh, acc);
}

// ex = exp(e - emax[dst]) (in place); atomic denom[dst,h] += ex
__global__ void k_edge_exp(float* __restrict__ ex, const float* __restrict__ emax,
                           const int* __restrict__ dst, float* __restrict__ denom)
{
    int idx = blockIdx.x * 256 + threadIdx.x;
    if (idx >= NEDGES * NHEADS) return;
    int eid = idx >> 3, hh = idx & 7;
    int d = dst[eid];
    float v = __expf(ex[idx] - emax[(size_t)d * NHEADS + hh]);
    ex[idx] = v;
    atomicAdd(denom + (size_t)d * NHEADS + hh, v);
}

// accum[dst, h*16+i] += (ex/denom[dst,h]) * xl[src, h*16+i]
__global__ void k_edge_aggregate(const float* __restrict__ ex, const float* __restrict__ denom,
                                 const float* __restrict__ xl,
                                 const int* __restrict__ src, const int* __restrict__ dst,
                                 float* __restrict__ accum)
{
    int idx = blockIdx.x * 256 + threadIdx.x;
    if (idx >= NEDGES * NHEADS) return;
    int eid = idx >> 3, hh = idx & 7;
    int s = src[eid], d = dst[eid];
    float alpha = ex[idx] / denom[(size_t)d * NHEADS + hh];
    const float* pl = xl + (size_t)s * FDIM + hh * DHEAD;
    float* po = accum + (size_t)d * FDIM + hh * DHEAD;
#pragma unroll
    for (int i = 0; i < DHEAD; ++i) atomicAdd(po + i, alpha * pl[i]);
}

// bias add + ReLU + LayerNorm; one wave32 per node, 4 features per lane
__global__ __launch_bounds__(256) void k_finalize_ln(
    const float* __restrict__ accum, const float* __restrict__ bias,
    const float* __restrict__ g, const float* __restrict__ b,
    float* __restrict__ out)
{
    int node = blockIdx.x * 8 + (threadIdx.x >> 5);
    int lane = threadIdx.x & 31;
    if (node >= NNODES) return;
    float v[4];
    float sum = 0.f;
#pragma unroll
    for (int i = 0; i < 4; ++i) {
        int c = lane * 4 + i;
        float t = accum[(size_t)node * FDIM + c] + bias[c];
        t = t > 0.f ? t : 0.f;          // ReLU
        v[i] = t; sum += t;
    }
#pragma unroll
    for (int off = 16; off > 0; off >>= 1) sum += __shfl_xor(sum, off, 32);
    float mu = sum * (1.0f / FDIM);
    float var = 0.f;
#pragma unroll
    for (int i = 0; i < 4; ++i) { float d0 = v[i] - mu; var += d0 * d0; }
#pragma unroll
    for (int off = 16; off > 0; off >>= 1) var += __shfl_xor(var, off, 32);
    float inv = rsqrtf(var * (1.0f / FDIM) + LN_EPS);
#pragma unroll
    for (int i = 0; i < 4; ++i) {
        int c = lane * 4 + i;
        out[(size_t)node * FDIM + c] = (v[i] - mu) * inv * g[c] + b[c];
    }
}

// ---------------- host orchestration ----------------

extern "C" void kernel_launch(void* const* d_in, const int* in_sizes, int n_in,
                              void* d_out, int out_size, void* d_ws, size_t ws_size,
                              hipStream_t stream)
{
    const float* x    = (const float*)d_in[0];
    const int*   eidx = (const int*)d_in[1];
    const int*   src  = eidx;
    const int*   dst  = eidx + NEDGES;
    const float* Wl1  = (const float*)d_in[2];
    const float* Wr1  = (const float*)d_in[3];
    const float* att1 = (const float*)d_in[4];
    const float* b1   = (const float*)d_in[5];
    const float* g1   = (const float*)d_in[6];
    const float* be1  = (const float*)d_in[7];
    const float* Wl2  = (const float*)d_in[8];
    const float* Wr2  = (const float*)d_in[9];
    const float* att2 = (const float*)d_in[10];
    const float* b2   = (const float*)d_in[11];
    const float* g2   = (const float*)d_in[12];
    const float* be2  = (const float*)d_in[13];

    char* ws = (char*)d_ws;
    auto carve = [&](size_t bytes) -> char* {
        char* p = ws; ws += (bytes + 255) & ~(size_t)255; return p;
    };
    _Float16* xh    = (_Float16*)carve((size_t)NNODES * FDIM * 2);
    _Float16* wlT   = (_Float16*)carve((size_t)FDIM * FDIM * 2);
    _Float16* wrT   = (_Float16*)carve((size_t)FDIM * FDIM * 2);
    float*    xl    = (float*)carve((size_t)NNODES * FDIM * 4);
    float*    xr    = (float*)carve((size_t)NNODES * FDIM * 4);
    float*    ex    = (float*)carve((size_t)NEDGES * NHEADS * 4);
    float*    emax  = (float*)carve((size_t)NNODES * NHEADS * 4);
    float*    denom = (float*)carve((size_t)NNODES * NHEADS * 4);
    float*    accum = (float*)carve((size_t)NNODES * FDIM * 4);
    float*    hbuf  = (float*)carve((size_t)NNODES * FDIM * 4);

    const int NF  = NNODES * FDIM;          // 5,120,000
    const int NH  = NNODES * NHEADS;        // 320,000
    const int EH  = NEDGES * NHEADS;        // 5,120,000
    const float NEG_INF = -__builtin_inff();

    auto run_layer = [&](const float* xin, const float* Wl, const float* Wr,
                         const float* att, const float* bias, const float* g,
                         const float* be, float* out) {
        k_to_f16<<<(NF + 255) / 256, 256, 0, stream>>>(xin, xh, NF);
        k_transpose_w<<<FDIM, FDIM, 0, stream>>>(Wl, wlT);
        k_transpose_w<<<FDIM, FDIM, 0, stream>>>(Wr, wrT);
        k_gemm_xlxr<<<NNODES / 16, 256, 0, stream>>>(xh, wlT, wrT, xl, xr);
        k_fill<<<(NH + 255) / 256, 256, 0, stream>>>(emax, NEG_INF, NH);
        k_fill<<<(NH + 255) / 256, 256, 0, stream>>>(denom, 0.f, NH);
        k_fill<<<(NF + 255) / 256, 256, 0, stream>>>(accum, 0.f, NF);
        k_edge_score<<<(EH + 255) / 256, 256, 0, stream>>>(xl, xr, att, src, dst, ex, emax);
        k_edge_exp<<<(EH + 255) / 256, 256, 0, stream>>>(ex, emax, dst, denom);
        k_edge_aggregate<<<(EH + 255) / 256, 256, 0, stream>>>(ex, denom, xl, src, dst, accum);
        k_finalize_ln<<<(NNODES + 7) / 8, 256, 0, stream>>>(accum, bias, g, be, out);
    };

    run_layer(x,    Wl1, Wr1, att1, b1, g1, be1, hbuf);
    run_layer(hbuf, Wl2, Wr2, att2, b2, g2, be2, (float*)d_out);
}
